// ComparatorNBit_23407571764174
// MI455X (gfx1250) — matrix operand order repaired and checked by reading, not compiled, not run on hardware
//
#include <hip/hip_runtime.h>

typedef __attribute__((ext_vector_type(16))) _Float16 v16h;
typedef __attribute__((ext_vector_type(8)))  float    v8f;
typedef __attribute__((ext_vector_type(4)))  float    f32x4;

// Per 16-row tile (one wave):
//   diff = A - B elementwise (exact in {-1,0,1})
//   acc  = diff(16x32) * W(32x16)   -- single v_wmma_f32_16x16x32_f16
// W col0: 2^(15-k) for k<16 (hi half), col1: 2^(31-k) for k>=16 (lo half).
// acc col0 = hiA-hiB, col1 = loA-loB (exact integers, |.| <= 65535).
// Combine: d = fma(dlo, 2^-16, dhi)  => sign(d)/d==0 give exact 32-bit compare.
__global__ __launch_bounds__(256) void cmp32_wmma_kernel(
    const float* __restrict__ A, const float* __restrict__ B,
    float* __restrict__ outGt, float* __restrict__ outEq, int nrows)
{
    const int lane = threadIdx.x & 31;
    const int wave = threadIdx.x >> 5;

    const long ntiles = ((long)nrows + 15) / 16;
    const long step   = (long)gridDim.x * 8;
    long tile         = (long)blockIdx.x * 8 + wave;

    const int n  = lane & 15;                  // B/D column owned by this lane
    const int kb = (lane < 16) ? 0 : 16;       // unused; B operand uses the k0 chunk order too
    const int k0 = (lane < 16) ? 0 : 8;        // A-operand K-chunk for this lane group

    // Weights in the 16-bit B-matrix 32x16 layout: lane L -> col L%16,
    // half elem h -> K = k0 + h (h<8), K = k0 + 16 + (h-8) (h>=8).
    v16h wp;
#pragma unroll
    for (int h = 0; h < 16; ++h) {
        const int K = (h < 8) ? (k0 + h) : (k0 + 8 + h);
        float w = 0.0f;
        if (n == 0 && K < 16)  w = (float)(1u << (15 - K));   // hi half, MSB first
        if (n == 1 && K >= 16) w = (float)(1u << (31 - K));   // lo half
        wp[h] = (_Float16)w;
    }

    for (; tile < ntiles; tile += step) {
        const long base = tile * 16;

        // A-operand layout (16-bit A 16x32): lane L -> row M = L%16,
        // lanes 0-15 hold K = 0..7 & 16..23, lanes 16-31 hold K = 8..15 & 24..31.
        long row = base + n;
        if (row >= nrows) row = nrows - 1;     // clamp loads on partial tile

        const float* rA = A + row * 32 + k0;
        const float* rB = B + row * 32 + k0;

        // Streaming (512 MB >> 192 MB L2): non-temporal b128 loads.
        f32x4 a0 = __builtin_nontemporal_load((const f32x4*)(rA));
        f32x4 a1 = __builtin_nontemporal_load((const f32x4*)(rA + 4));
        f32x4 a2 = __builtin_nontemporal_load((const f32x4*)(rA + 16));
        f32x4 a3 = __builtin_nontemporal_load((const f32x4*)(rA + 20));
        f32x4 b0 = __builtin_nontemporal_load((const f32x4*)(rB));
        f32x4 b1 = __builtin_nontemporal_load((const f32x4*)(rB + 4));
        f32x4 b2 = __builtin_nontemporal_load((const f32x4*)(rB + 16));
        f32x4 b3 = __builtin_nontemporal_load((const f32x4*)(rB + 20));

        // diff in f32 (exact), pack into v16h: half elem h -> K = k0+h (h<8),
        // K = k0+16+(h-8) (h>=8).
        v16h dv;
#pragma unroll
        for (int i = 0; i < 4; ++i) {
            dv[i]      = (_Float16)(a0[i] - b0[i]);
            dv[4 + i]  = (_Float16)(a1[i] - b1[i]);
            dv[8 + i]  = (_Float16)(a2[i] - b2[i]);
            dv[12 + i] = (_Float16)(a3[i] - b3[i]);
        }

        v8f acc = {};
        acc = __builtin_amdgcn_wmma_f32_16x16x32_f16(false, dv, false, wp,
                                                     (short)0, acc, false, false);

        // D layout: lane L holds column L%16, rows (L/16)*8 + v in acc[v].
        // hi diff on lanes 0/16 (col 0), lo diff on lanes 1/17 (col 1).
        float gt[8], eq[8];
#pragma unroll
        for (int r = 0; r < 8; ++r) {
            const float dlo = __shfl_xor(acc[r], 1, 32);     // lanes 0/16 get col 1
            const float d   = __builtin_fmaf(dlo, 0x1p-16f, acc[r]);
            gt[r] = (d > 0.0f)  ? 1.0f : 0.0f;
            eq[r] = (d == 0.0f) ? 1.0f : 0.0f;
        }

        // Lanes 0 and 16 each own 8 consecutive rows -> coalesced b128 NT stores.
        if (n == 0) {
            const long r0 = base + (lane >> 4) * 8;
            if (r0 + 8 <= nrows) {
                __builtin_nontemporal_store(f32x4{gt[0], gt[1], gt[2], gt[3]}, (f32x4*)(outGt + r0));
                __builtin_nontemporal_store(f32x4{gt[4], gt[5], gt[6], gt[7]}, (f32x4*)(outGt + r0 + 4));
                __builtin_nontemporal_store(f32x4{eq[0], eq[1], eq[2], eq[3]}, (f32x4*)(outEq + r0));
                __builtin_nontemporal_store(f32x4{eq[4], eq[5], eq[6], eq[7]}, (f32x4*)(outEq + r0 + 4));
            } else {
                for (int r = 0; r < 8; ++r) {
                    if (r0 + r < nrows) { outGt[r0 + r] = gt[r]; outEq[r0 + r] = eq[r]; }
                }
            }
        }
    }
}

extern "C" void kernel_launch(void* const* d_in, const int* in_sizes, int n_in,
                              void* d_out, int out_size, void* d_ws, size_t ws_size,
                              hipStream_t stream) {
    const float* A = (const float*)d_in[0];
    const float* B = (const float*)d_in[1];
    float* out = (float*)d_out;

    const int  nrows  = in_sizes[0] / 32;          // 2,000,000
    const long ntiles = ((long)nrows + 15) / 16;   // 16 rows per wave-tile
    long blocks = (ntiles + 7) / 8;                // 8 waves per 256-thread block
    if (blocks > 2048) blocks = 2048;              // grid-stride over tiles

    cmp32_wmma_kernel<<<(int)blocks, 256, 0, stream>>>(A, B, out, out + nrows, nrows);
}
